// SoftKMeans_31430570672817
// MI455X (gfx1250) — matrix operand order repaired and checked by reading, not compiled, not run on hardware
//
#include <hip/hip_runtime.h>
#include <hip/hip_bf16.h>

typedef __attribute__((ext_vector_type(16))) __bf16 v16bf;
typedef __attribute__((ext_vector_type(8)))  float  v8f;

#define D_DIM  256
#define K_CENT 1024

// ---------------------------------------------------------------------------
// Prep: centroids fp32 -> bf16 (for WMMA), exact fp32 ||c||^2, zero loss slot.
// One block per centroid row (1024 blocks x 256 threads). ~1 MB of traffic.
// ---------------------------------------------------------------------------
__global__ __launch_bounds__(256) void skm_prep(const float* __restrict__ cent,
                                                __bf16* __restrict__ cbf,
                                                float* __restrict__ c2,
                                                float* __restrict__ loss_slot) {
  const int k = blockIdx.x;
  const int t = threadIdx.x;
  const float v = cent[(size_t)k * D_DIM + t];
  cbf[(size_t)k * D_DIM + t] = (__bf16)v;

  float s = v * v;
  #pragma unroll
  for (int m = 1; m < 32; m <<= 1) s += __shfl_xor(s, m, 32);

  __shared__ float ls[8];
  if ((t & 31) == 0) ls[t >> 5] = s;
  __syncthreads();
  if (t == 0) {
    float tot = 0.f;
    #pragma unroll
    for (int i = 0; i < 8; ++i) tot += ls[i];
    c2[k] = tot;
    if (blockIdx.x == 0) *loss_slot = 0.f;
  }
}

// ---------------------------------------------------------------------------
// Fused (x.c^T) GEMM via v_wmma_f32_16x16x32_bf16 + softmax(-dist) + loss.
// Block = 8 waves: row-groups g=wid/4 (16 rows each), col-quadrant w4=wid%4
// (256 centroids each). Each wave: 16 tiles of 16x16 f32 accumulators.
// ---------------------------------------------------------------------------
__global__ __launch_bounds__(256) void skm_main(const float* __restrict__ x,
                                                const __bf16* __restrict__ cbf,
                                                const float* __restrict__ c2,
                                                float* __restrict__ out,
                                                float* __restrict__ loss) {
  const int tid  = threadIdx.x;
  const int lane = tid & 31;
  const int wid  = tid >> 5;
  const int l15  = lane & 15;
  const int half = lane >> 4;          // 0: M/K low half, 1: high half
  const int g    = wid >> 2;           // row group (0/1)
  const int w4   = wid & 3;            // centroid quadrant (0..3)

  const int row0  = blockIdx.x * 32 + g * 16;  // first of this wave's 16 rows
  const int myrow = row0 + l15;                // row whose A data this lane loads
  const int n0    = w4 * 256;                  // first centroid column

  // ||c||^2 for the 16 columns this lane touches (col = n0 + t*16 + l15)
  float c2v[16];
  #pragma unroll
  for (int t = 0; t < 16; ++t) c2v[t] = c2[n0 + t * 16 + l15];

  v8f acc[16];
  #pragma unroll
  for (int t = 0; t < 16; ++t)
    #pragma unroll
    for (int j = 0; j < 8; ++j) acc[t][j] = 0.f;

  // A-fragment lane K offsets (contiguous, matching B):
  // lanes 0-15: K = kb+{0..15};  lanes 16-31: K = kb+{16..31}
  const int o    = half * 16;  // A fp32 offset
  const int koff = half * 16;  // B bf16 offset (lanes 0-15: K 0..15; 16-31: K 16..31)

  const float* xrow = x + (size_t)myrow * D_DIM;
  float xsq = 0.f;

  for (int kb = 0; kb < D_DIM; kb += 32) {
    const float* ap = xrow + kb + o;
    const float4 p0 = *(const float4*)(ap + 0);
    const float4 p1 = *(const float4*)(ap + 4);
    const float4 p2 = *(const float4*)(ap + 8);
    const float4 p3 = *(const float4*)(ap + 12);

    xsq += p0.x*p0.x + p0.y*p0.y + p0.z*p0.z + p0.w*p0.w
         + p1.x*p1.x + p1.y*p1.y + p1.z*p1.z + p1.w*p1.w
         + p2.x*p2.x + p2.y*p2.y + p2.z*p2.z + p2.w*p2.w
         + p3.x*p3.x + p3.y*p3.y + p3.z*p3.z + p3.w*p3.w;

    const v16bf a = { (__bf16)p0.x, (__bf16)p0.y, (__bf16)p0.z, (__bf16)p0.w,
                      (__bf16)p1.x, (__bf16)p1.y, (__bf16)p1.z, (__bf16)p1.w,
                      (__bf16)p2.x, (__bf16)p2.y, (__bf16)p2.z, (__bf16)p2.w,
                      (__bf16)p3.x, (__bf16)p3.y, (__bf16)p3.z, (__bf16)p3.w };

    #pragma unroll
    for (int t = 0; t < 16; ++t) {
      // B column n = centroid row n, contiguous bf16 K-chunk per lane (32B).
      const v16bf b = *(const v16bf*)(cbf + (size_t)(n0 + t * 16 + l15) * D_DIM + kb + koff);
      acc[t] = __builtin_amdgcn_wmma_f32_16x16x32_bf16(
          /*neg_a=*/false, a, /*neg_b=*/false, b,
          /*c_mod=*/(short)0, acc[t], /*reuse_a=*/false, /*reuse_b=*/false);
    }
  }

  // Exact fp32 ||x||^2 per row: lanes l and l+16 together covered all 256 K.
  const float x2c = xsq + __shfl_xor(xsq, 16, 32);  // lane m & m+16 hold row m
  float xv[8];
  #pragma unroll
  for (int j = 0; j < 8; ++j) xv[j] = __shfl(x2c, half * 8 + j, 32);

  // acc -> dist = sqrt(max(x2 + c2 - 2*xc, 0))
  #pragma unroll
  for (int t = 0; t < 16; ++t)
    #pragma unroll
    for (int j = 0; j < 8; ++j) {
      const float sq = fmaxf(xv[j] + c2v[t] - 2.0f * acc[t][j], 0.0f);
      acc[t][j] = __builtin_sqrtf(sq);
    }

  __shared__ float lmin[2][4][16];
  __shared__ float lse [2][4][16];
  __shared__ float lses[2][4][16];

  // row-min(dist) over this wave's 256 cols: tiles, then half-wave butterfly
  float dmin[8];
  #pragma unroll
  for (int j = 0; j < 8; ++j) dmin[j] = 3.402823e38f;
  #pragma unroll
  for (int t = 0; t < 16; ++t)
    #pragma unroll
    for (int j = 0; j < 8; ++j) dmin[j] = fminf(dmin[j], acc[t][j]);
  #pragma unroll
  for (int j = 0; j < 8; ++j) {
    float v = dmin[j];
    #pragma unroll
    for (int m = 1; m < 16; m <<= 1) v = fminf(v, __shfl_xor(v, m, 32));
    dmin[j] = v;
  }
  if (l15 == 0) {  // lanes 0 (rows 0-7) and 16 (rows 8-15)
    #pragma unroll
    for (int j = 0; j < 8; ++j) lmin[g][w4][half * 8 + j] = dmin[j];
  }
  __syncthreads();

  float rmin[8];
  #pragma unroll
  for (int j = 0; j < 8; ++j) {
    const int jj = half * 8 + j;
    rmin[j] = fminf(fminf(lmin[g][0][jj], lmin[g][1][jj]),
                    fminf(lmin[g][2][jj], lmin[g][3][jj]));
  }

  // e = exp(dmin - dist); accumulate sum(e) and sum(e*sq) per row
  float se[8], ses[8];
  #pragma unroll
  for (int j = 0; j < 8; ++j) { se[j] = 0.f; ses[j] = 0.f; }
  #pragma unroll
  for (int t = 0; t < 16; ++t)
    #pragma unroll
    for (int j = 0; j < 8; ++j) {
      const float d = acc[t][j];
      const float e = __expf(rmin[j] - d);
      acc[t][j] = e;
      se[j]  += e;
      ses[j] += e * d * d;
    }
  #pragma unroll
  for (int j = 0; j < 8; ++j) {
    float a = se[j], b = ses[j];
    #pragma unroll
    for (int m = 1; m < 16; m <<= 1) {
      a += __shfl_xor(a, m, 32);
      b += __shfl_xor(b, m, 32);
    }
    se[j] = a; ses[j] = b;
  }
  if (l15 == 0) {
    #pragma unroll
    for (int j = 0; j < 8; ++j) {
      lse [g][w4][half * 8 + j] = se[j];
      lses[g][w4][half * 8 + j] = ses[j];
    }
  }
  __syncthreads();

  float inv[8];
  float lossp = 0.f;
  #pragma unroll
  for (int j = 0; j < 8; ++j) {
    const int jj = half * 8 + j;
    const float Se = lse [g][0][jj] + lse [g][1][jj] + lse [g][2][jj] + lse [g][3][jj];
    const float Ss = lses[g][0][jj] + lses[g][1][jj] + lses[g][2][jj] + lses[g][3][jj];
    const float r  = __builtin_amdgcn_rcpf(Se);
    inv[j] = r;
    lossp += Ss * r;   // sum(soft*sq) for this row
  }

  // one loss contribution per 16-row group: lanes 0 (rows 0-7) + 16 (rows 8-15)
  if (w4 == 0 && l15 == 0) {
    lossp += __shfl_xor(lossp, 16, 32);
    if (lane == 0) atomicAdd(loss, lossp * (1.0f / 16.0f));
  }

  // soft = e / sum(e)
  #pragma unroll
  for (int t = 0; t < 16; ++t) {
    const size_t col = (size_t)(n0 + t * 16 + l15);
    #pragma unroll
    for (int j = 0; j < 8; ++j) {
      const size_t row = (size_t)(row0 + half * 8 + j);
      out[row * K_CENT + col] = acc[t][j] * inv[j];
    }
  }
}

// ---------------------------------------------------------------------------
extern "C" void kernel_launch(void* const* d_in, const int* in_sizes, int n_in,
                              void* d_out, int out_size, void* d_ws, size_t ws_size,
                              hipStream_t stream) {
  const float* x    = (const float*)d_in[0];
  const float* cent = (const float*)d_in[1];

  __bf16* cbf = (__bf16*)d_ws;                                            // 512 KB
  float*  c2  = (float*)((char*)d_ws + (size_t)K_CENT * D_DIM * sizeof(__bf16)); // 4 KB

  float* out  = (float*)d_out;
  float* loss = out + (out_size - 1);

  const int rows = in_sizes[0] / D_DIM;  // B*N = 131072

  skm_prep<<<K_CENT, 256, 0, stream>>>(cent, cbf, c2, loss);
  skm_main<<<rows / 32, 256, 0, stream>>>(x, cbf, c2, out, loss);
}
